// HeteroGNN_34153579938035
// MI455X (gfx1250) — matrix operand order repaired and checked by reading, not compiled, not run on hardware
//
#include <hip/hip_runtime.h>
#include <stdint.h>

#define NN   16384
#define CC   128
#define TOPK 20

typedef __bf16 bf16_t;
typedef bf16_t v16bf __attribute__((ext_vector_type(16)));
typedef float  v8f   __attribute__((ext_vector_type(8)));

struct Frag { unsigned u[8]; };

__device__ __forceinline__ v8f wmma_bf16f32(const Frag& a, const Frag& b, v8f c) {
    union U { Frag f; v16bf v; };
    U ua; ua.f = a;
    U ub; ub.f = b;
    return __builtin_amdgcn_wmma_f32_16x16x32_bf16(false, ua.v, false, ub.v,
                                                   (short)0, c, false, false);
}

// A fragment (16x32 bf16, MxK): lane m holds row m; lane group h=lane>>4.
// VGPR v in 0..3 : K = h*8 + 2v       -> dword  kk_dw + h*4 + v
// VGPR v in 4..7 : K = 16 + h*8 + 2v' -> dword  kk_dw + 8 + h*4 + v'
__device__ __forceinline__ void load_a_frag(const unsigned* row, int kk_dw, int h, Frag& f) {
    int b0 = kk_dw + h * 4;
#pragma unroll
    for (int v = 0; v < 4; ++v) f.u[v] = row[b0 + v];
#pragma unroll
    for (int v = 0; v < 4; ++v) f.u[4 + v] = row[b0 + 8 + v];
}

// B fragment (32x16 bf16, KxN): lane n holds column n; same K order as A fragment
__device__ __forceinline__ void load_b_frag(const unsigned* row, int kk_dw, int h, Frag& f) {
    int b0 = kk_dw + h * 4;
#pragma unroll
    for (int v = 0; v < 4; ++v) f.u[v] = row[b0 + v];
#pragma unroll
    for (int v = 0; v < 4; ++v) f.u[4 + v] = row[b0 + 8 + v];
}

__device__ __forceinline__ unsigned short f2bf(float x) {
    unsigned u = __float_as_uint(x);
    unsigned lsb = (u >> 16) & 1u;
    u += 0x7fffu + lsb;
    return (unsigned short)(u >> 16);
}

// ---------------------------------------------------------------- conversion
__global__ void cvt_f32_bf16(const float* __restrict__ s, unsigned short* __restrict__ d, int n) {
    int i = blockIdx.x * blockDim.x + threadIdx.x;
    int stride = gridDim.x * blockDim.x;
    for (; i < n; i += stride) d[i] = f2bf(s[i]);
}

// ---------------------------------------------------------------- q/k GEMMs
__global__ void qk_gemm_kernel(const unsigned short* __restrict__ xb,
                               const unsigned short* __restrict__ wqb,
                               const unsigned short* __restrict__ wkb,
                               const float* __restrict__ bq,
                               const float* __restrict__ bk,
                               unsigned short* __restrict__ qb,
                               unsigned short* __restrict__ kb) {
    const int wave = threadIdx.x >> 5;
    const int lane = threadIdx.x & 31;
    const int m = lane & 15, h = lane >> 4;
    const int tile = blockIdx.x * 4 + wave;     // 16384 wave-tiles
    const int mat  = tile >> 13;                // 0 = q, 1 = k
    const int rem  = tile & 8191;
    const int rowt = rem >> 3;
    const int colt = rem & 7;

    const unsigned short* W = mat ? wkb : wqb;
    const float* bias       = mat ? bk : bq;
    unsigned short* out     = mat ? kb : qb;

    const unsigned* xrow = (const unsigned*)xb + (rowt * 16 + m) * 64;
    const unsigned* wrow = (const unsigned*)W  + (colt * 16 + m) * 64;

    v8f c = {0.f,0.f,0.f,0.f,0.f,0.f,0.f,0.f};
#pragma unroll
    for (int f = 0; f < 4; ++f) {
        Frag a, b;
        load_a_frag(xrow, f * 16, h, a);
        load_b_frag(wrow, f * 16, h, b);
        c = wmma_bf16f32(a, b, c);
    }
    float bv = bias[colt * 16 + m];
#pragma unroll
    for (int r = 0; r < 8; ++r)
        out[(rowt * 16 + r + 8 * h) * CC + colt * 16 + m] = f2bf(c[r] + bv);
}

// ---------------------------------------------------------------- sim + topk
// sim = k @ q^T, fused running top-20 per row, then softmax.
// Each wave: 2 column tiles / iteration -> 2 independent WMMA chains
// (fills bf16 WMMA->WMMA hazard slots), and all 32 lanes run the top-k scan
// (lane group h scans tile h).
__global__ void sim_topk_kernel(const unsigned short* __restrict__ kb,
                                const unsigned short* __restrict__ qb,
                                float* __restrict__ score_out,
                                int* __restrict__ idx_out) {
    __shared__ float stile[4 * 512];
    __shared__ float slv[4][32][TOPK];
    __shared__ int   sli[4][32][TOPK];
    const int wave = threadIdx.x >> 5;
    const int lane = threadIdx.x & 31;
    const int m = lane & 15, h = lane >> 4;
    const int rowt = blockIdx.x;
    const float NEG = -3.402823466e38f;

    const unsigned* krow = (const unsigned*)kb + (rowt * 16 + m) * 64;
    Frag afr[4];
#pragma unroll
    for (int f = 0; f < 4; ++f) load_a_frag(krow, f * 16, h, afr[f]);

    float topv[TOPK]; int topi[TOPK];
#pragma unroll
    for (int e = 0; e < TOPK; ++e) { topv[e] = NEG; topi[e] = 0; }
    float minv = NEG; int minslot = 0;

    float* mytile = stile + wave * 512;
    const unsigned* qb32 = (const unsigned*)qb;
    for (int colt0 = wave * 2; colt0 < NN / 16; colt0 += 8) {
        const unsigned* qrow0 = qb32 + ((colt0 + 0) * 16 + m) * 64;
        const unsigned* qrow1 = qb32 + ((colt0 + 1) * 16 + m) * 64;
        // hide L2 latency of the next iteration's q tiles
        __builtin_prefetch(qrow0 + 8 * 16 * 64, 0, 1);
        __builtin_prefetch(qrow1 + 8 * 16 * 64, 0, 1);
        v8f c0 = {0.f,0.f,0.f,0.f,0.f,0.f,0.f,0.f};
        v8f c1 = {0.f,0.f,0.f,0.f,0.f,0.f,0.f,0.f};
#pragma unroll
        for (int f = 0; f < 4; ++f) {
            Frag b0, b1;
            load_b_frag(qrow0, f * 16, h, b0);
            load_b_frag(qrow1, f * 16, h, b1);
            c0 = wmma_bf16f32(afr[f], b0, c0);     // two independent chains:
            c1 = wmma_bf16f32(afr[f], b1, c1);     // interleave fills hazard slots
        }
#pragma unroll
        for (int r = 0; r < 8; ++r) mytile[(r + 8 * h) * 16 + m]       = c0[r];
#pragma unroll
        for (int r = 0; r < 8; ++r) mytile[512/2 + (r + 8 * h) * 16 + m] = c1[r];
        // same-wave LDS write->read: in-order, no barrier needed
        {
            const float* tsel = mytile + h * 256 + m * 16;
            const int cbase = (colt0 + h) * 16;
            for (int cc = 0; cc < 16; ++cc) {
                float s = tsel[cc];
                if (s > minv) {
                    int ci = cbase + cc;
#pragma unroll
                    for (int e = 0; e < TOPK; ++e)
                        if (e == minslot) { topv[e] = s; topi[e] = ci; }
                    minv = topv[0]; minslot = 0;
#pragma unroll
                    for (int e = 1; e < TOPK; ++e)
                        if (topv[e] < minv) { minv = topv[e]; minslot = e; }
                }
            }
        }
    }
#pragma unroll
    for (int e = 0; e < TOPK; ++e) { slv[wave][lane][e] = topv[e]; sli[wave][lane][e] = topi[e]; }
    __syncthreads();
    if (wave == 0 && lane < 16) {
        const int row = lane;
        float outv[TOPK]; int outi[TOPK];
        for (int t = 0; t < TOPK; ++t) {
            float best = NEG; int bw = 0, bl = 0, be = 0;
            for (int w = 0; w < 4; ++w)
                for (int hf = 0; hf < 2; ++hf) {
                    int l = row + 16 * hf;
                    for (int e = 0; e < TOPK; ++e) {
                        float v = slv[w][l][e];
                        if (v > best) { best = v; bw = w; bl = l; be = e; }
                    }
                }
            outv[t] = best;
            outi[t] = sli[bw][bl][be];
            slv[bw][bl][be] = NEG;
        }
        float mx = outv[0];              // selection order is descending
        float ev[TOPK]; float esum = 0.f;
        for (int t = 0; t < TOPK; ++t) { float e = __expf(outv[t] - mx); ev[t] = e; esum += e; }
        float inv = 1.0f / esum;
        int grow = rowt * 16 + row;
        for (int t = 0; t < TOPK; ++t) {
            score_out[grow * TOPK + t] = ev[t] * inv;
            idx_out[grow * TOPK + t]   = outi[t];
        }
    }
}

// ---------------------------------------------------------------- message MLP + aggregate
__global__ void msg_mlp_kernel(const unsigned short* __restrict__ xb,
                               const unsigned short* __restrict__ wm1b,
                               const float* __restrict__ bm1,
                               const unsigned short* __restrict__ wm2b,
                               const float* __restrict__ bm2,
                               const float* __restrict__ score,
                               const int* __restrict__ nidx,
                               float* __restrict__ hagg) {
    __shared__ float shagg[8 * CC];
    __shared__ float ssc[4][16];
    __shared__ unsigned short shid[4 * 2048];
    const int tid = threadIdx.x;
    const int wave = tid >> 5, lane = tid & 31;
    const int m = lane & 15, h = lane >> 4;
    const int nodebase = blockIdx.x * 8;

    for (int i = tid; i < 8 * CC; i += 128) shagg[i] = 0.f;
    __syncthreads();

    unsigned short* hid = shid + wave * 2048;
    const unsigned* hid32 = (const unsigned*)hid;

    for (int t = wave; t < 10; t += 4) {           // 160 pairs = 10 row tiles
        int p  = t * 16 + m;
        int in = nodebase + p / 20;
        int kk = p - (p / 20) * 20;
        int jn = nidx[in * TOPK + kk];
        if (h == 0) ssc[wave][m] = score[in * TOPK + kk];
        const unsigned* xi = (const unsigned*)xb + in * 64;
        const unsigned* xj = (const unsigned*)xb + jn * 64;

#pragma unroll 2
        for (int ct = 0; ct < 8; ++ct) {           // hidden = relu(row @ Wm1^T + b)
            v8f c = {0.f,0.f,0.f,0.f,0.f,0.f,0.f,0.f};
            const unsigned* wrow = (const unsigned*)wm1b + (ct * 16 + m) * 128;
#pragma unroll
            for (int f = 0; f < 8; ++f) {          // K = 256
                Frag a, b;
                load_a_frag((f < 4) ? xi : xj, (f & 3) * 16, h, a);
                load_b_frag(wrow, f * 16, h, b);
                c = wmma_bf16f32(a, b, c);
            }
            float bv = bm1[ct * 16 + m];
#pragma unroll
            for (int r = 0; r < 8; ++r) {
                float vv = c[r] + bv;
                vv = vv > 0.f ? vv : 0.f;
                hid[(r + 8 * h) * CC + ct * 16 + m] = f2bf(vv);
            }
        }
#pragma unroll 2
        for (int ct2 = 0; ct2 < 8; ++ct2) {        // out = hidden @ Wm2^T + b
            v8f c2 = {0.f,0.f,0.f,0.f,0.f,0.f,0.f,0.f};
            const unsigned* wrow2 = (const unsigned*)wm2b + (ct2 * 16 + m) * 64;
#pragma unroll
            for (int f = 0; f < 4; ++f) {          // K = 128
                Frag a, b;
                load_a_frag(hid32 + m * 64, f * 16, h, a);
                load_b_frag(wrow2, f * 16, h, b);
                c2 = wmma_bf16f32(a, b, c2);
            }
            float bv2 = bm2[ct2 * 16 + m];
#pragma unroll
            for (int r = 0; r < 8; ++r) {
                int M = r + 8 * h;
                int pl = t * 16 + M;
                int li = pl / 20;                  // local node 0..7
                float val = (c2[r] + bv2) * ssc[wave][M];
                atomicAdd(&shagg[li * CC + ct2 * 16 + m], val);
            }
        }
    }
    __syncthreads();
    for (int i = tid; i < 8 * CC; i += 128) hagg[nodebase * CC + i] = shagg[i];
}

// ---------------------------------------------------------------- update MLP + residual
__global__ void upd_mlp_kernel(const float* __restrict__ x,
                               const unsigned short* __restrict__ xb,
                               const unsigned short* __restrict__ haggb,
                               const unsigned short* __restrict__ wu1b,
                               const float* __restrict__ bu1,
                               const unsigned short* __restrict__ wu2b,
                               const float* __restrict__ bu2,
                               float* __restrict__ out) {
    __shared__ unsigned short shid[4 * 2048];
    const int wave = threadIdx.x >> 5, lane = threadIdx.x & 31;
    const int m = lane & 15, h = lane >> 4;
    const int tile = blockIdx.x * 4 + wave;        // 1024 tiles
    const int row0 = tile * 16;

    unsigned short* hid = shid + wave * 2048;
    const unsigned* hid32 = (const unsigned*)hid;
    const unsigned* xi = (const unsigned*)xb    + (row0 + m) * 64;
    const unsigned* hj = (const unsigned*)haggb + (row0 + m) * 64;

#pragma unroll 2
    for (int ct = 0; ct < 8; ++ct) {
        v8f c = {0.f,0.f,0.f,0.f,0.f,0.f,0.f,0.f};
        const unsigned* wrow = (const unsigned*)wu1b + (ct * 16 + m) * 128;
#pragma unroll
        for (int f = 0; f < 8; ++f) {
            Frag a, b;
            load_a_frag((f < 4) ? xi : hj, (f & 3) * 16, h, a);
            load_b_frag(wrow, f * 16, h, b);
            c = wmma_bf16f32(a, b, c);
        }
        float bv = bu1[ct * 16 + m];
#pragma unroll
        for (int r = 0; r < 8; ++r) {
            float vv = c[r] + bv;
            vv = vv > 0.f ? vv : 0.f;
            hid[(r + 8 * h) * CC + ct * 16 + m] = f2bf(vv);
        }
    }
#pragma unroll 2
    for (int ct2 = 0; ct2 < 8; ++ct2) {
        v8f c2 = {0.f,0.f,0.f,0.f,0.f,0.f,0.f,0.f};
        const unsigned* wrow2 = (const unsigned*)wu2b + (ct2 * 16 + m) * 64;
#pragma unroll
        for (int f = 0; f < 4; ++f) {
            Frag a, b;
            load_a_frag(hid32 + m * 64, f * 16, h, a);
            load_b_frag(wrow2, f * 16, h, b);
            c2 = wmma_bf16f32(a, b, c2);
        }
        float bv2 = bu2[ct2 * 16 + m];
#pragma unroll
        for (int r = 0; r < 8; ++r) {
            int gi = (row0 + r + 8 * h) * CC + ct2 * 16 + m;
            out[gi] = x[gi] + c2[r] + bv2;
        }
    }
}

// ---------------------------------------------------------------- launcher
extern "C" void kernel_launch(void* const* d_in, const int* in_sizes, int n_in,
                              void* d_out, int out_size, void* d_ws, size_t ws_size,
                              hipStream_t stream) {
    (void)in_sizes; (void)n_in; (void)out_size; (void)ws_size;
    const float* x   = (const float*)d_in[0];
    const float* Wq  = (const float*)d_in[1];
    const float* bq  = (const float*)d_in[2];
    const float* Wk  = (const float*)d_in[3];
    const float* bk  = (const float*)d_in[4];
    const float* Wm1 = (const float*)d_in[5];
    const float* bm1 = (const float*)d_in[6];
    const float* Wm2 = (const float*)d_in[7];
    const float* bm2 = (const float*)d_in[8];
    const float* Wu1 = (const float*)d_in[9];
    const float* bu1 = (const float*)d_in[10];
    const float* Wu2 = (const float*)d_in[11];
    const float* bu2 = (const float*)d_in[12];

    char* ws = (char*)d_ws;
    const size_t MB = 1024u * 1024u;
    unsigned short* xb    = (unsigned short*)(ws);             // 4 MB
    unsigned short* qb    = (unsigned short*)(ws + 4 * MB);    // 4 MB
    unsigned short* kb    = (unsigned short*)(ws + 8 * MB);    // 4 MB
    float*          hagg  = (float*)         (ws + 12 * MB);   // 8 MB
    unsigned short* haggb = (unsigned short*)(ws + 20 * MB);   // 4 MB
    float*          score = (float*)         (ws + 24 * MB);   // 1.25 MB
    int*            nidx  = (int*)           (ws + 26 * MB);   // 1.25 MB
    unsigned short* wqb   = (unsigned short*)(ws + 28 * MB);
    unsigned short* wkb   = wqb  + CC * CC;
    unsigned short* wm1b  = wkb  + CC * CC;
    unsigned short* wm2b  = wm1b + CC * 2 * CC;
    unsigned short* wu1b  = wm2b + CC * CC;
    unsigned short* wu2b  = wu1b + CC * 2 * CC;

    auto cvt = [&](const float* s, unsigned short* d, int n) {
        cvt_f32_bf16<<<(n + 255) / 256, 256, 0, stream>>>(s, d, n);
    };
    cvt(x,   xb,   NN * CC);
    cvt(Wq,  wqb,  CC * CC);
    cvt(Wk,  wkb,  CC * CC);
    cvt(Wm1, wm1b, CC * 2 * CC);
    cvt(Wm2, wm2b, CC * CC);
    cvt(Wu1, wu1b, CC * 2 * CC);
    cvt(Wu2, wu2b, CC * CC);

    qk_gemm_kernel <<<4096, 128, 0, stream>>>(xb, wqb, wkb, bq, bk, qb, kb);
    sim_topk_kernel<<<NN / 16, 128, 0, stream>>>(kb, qb, score, nidx);
    msg_mlp_kernel <<<NN / 8, 128, 0, stream>>>(xb, wm1b, bm1, wm2b, bm2, score, nidx, hagg);
    cvt(hagg, haggb, NN * CC);
    upd_mlp_kernel <<<NN / 64, 128, 0, stream>>>(x, xb, haggb, wu1b, bu1, wu2b, bu2, (float*)d_out);
}
